// MoERoutingLayer_58720792871362
// MI455X (gfx1250) — compile-verified, matrix operand order checked by reading, not run on hardware
//
#include <hip/hip_runtime.h>
#include <hip/hip_bf16.h>

// ---------------------------------------------------------------------------
// MoE routing layer, MI455X (gfx1250, wave32, WMMA).
// Strategy: mix expert conv weights by softmax routing weights FIRST (linear),
// then one per-batch implicit-GEMM 3x3 conv using v_wmma_f32_16x16x32_bf16
// with double-buffered LDS staging.
// ---------------------------------------------------------------------------

#define B_    32
#define CIN   64
#define COUT  64
#define H_    58
#define W_    58
#define OH    56
#define OW    56
#define NPIX  (OH*OW)        // 3136
#define RSIZE 512
#define NB    10
#define EDIM  64
#define KDIM  (CIN*9)        // 576
#define LDSTR 40             // padded LDS row stride (ushorts): 80B, 16B aligned

typedef __attribute__((ext_vector_type(16))) __bf16 v16bf;
typedef __attribute__((ext_vector_type(8)))  float  v8f;
typedef __attribute__((ext_vector_type(8)))  unsigned short u16x8;

__device__ __forceinline__ unsigned short f2bf(float f) {
    unsigned u = __builtin_bit_cast(unsigned, f);
    unsigned r = u + 0x7FFFu + ((u >> 16) & 1u);   // round-to-nearest-even
    return (unsigned short)(r >> 16);
}

__device__ __forceinline__ v16bf load_frag(const unsigned short* p) {
    // fragment = two contiguous 16B chunks: k in [h*8, h*8+8) and [16+h*8, ...)
    union { u16x8 u[2]; v16bf bf; } cvt;
    cvt.u[0] = *(const u16x8*)(p);
    cvt.u[1] = *(const u16x8*)(p + 16);
    return cvt.bf;
}

// ------------------------- kernel 1: routing -------------------------------
__global__ __launch_bounds__(128) void routing_kernel(
    const float* __restrict__ rv,  const float* __restrict__ W1,
    const float* __restrict__ b1,  const float* __restrict__ W2,
    const float* __restrict__ b2,  const float* __restrict__ emb,
    float* __restrict__ weights)          // [B_][NB]
{
    __shared__ float s_rv[RSIZE];
    __shared__ float s_h[128];
    __shared__ float s_r[EDIM];
    __shared__ float s_sim[NB];
    const int b = blockIdx.x, t = threadIdx.x;

    for (int i = t; i < RSIZE; i += 128) s_rv[i] = rv[b * RSIZE + i];
    __syncthreads();

    float acc = b1[t];
    for (int i = 0; i < RSIZE; ++i) acc += s_rv[i] * W1[i * 128 + t];
    s_h[t] = fmaxf(acc, 0.f);
    __syncthreads();

    if (t < EDIM) {
        float a2 = b2[t];
        for (int j = 0; j < 128; ++j) a2 += s_h[j] * W2[j * EDIM + t];
        s_r[t] = a2;
    }
    __syncthreads();

    if (t < NB) {
        float rn = 0.f;
        for (int j = 0; j < EDIM; ++j) rn += s_r[j] * s_r[j];
        rn = sqrtf(rn) + 1e-8f;
        float en = 0.f, dot = 0.f;
        for (int j = 0; j < EDIM; ++j) {
            float e = emb[t * EDIM + j];
            en += e * e; dot += s_r[j] * e;
        }
        en = sqrtf(en) + 1e-8f;
        s_sim[t] = dot / (rn * en);
    }
    __syncthreads();

    if (t == 0) {
        float mx = s_sim[0];
        for (int n = 1; n < NB; ++n) mx = fmaxf(mx, s_sim[n]);
        float ex[NB], sum = 0.f;
        for (int n = 0; n < NB; ++n) { ex[n] = __expf(s_sim[n] - mx); sum += ex[n]; }
        for (int n = 0; n < NB; ++n) weights[b * NB + n] = ex[n] / sum;
    }
}

// ----------------- kernel 2: mix expert weights -> bf16 A ------------------
__global__ __launch_bounds__(256) void mix_kernel(
    const float* __restrict__ conv_w,   // [NB][COUT][CIN][3][3]
    const float* __restrict__ conv_b,   // [NB][COUT]
    const float* __restrict__ weights,  // [B_][NB]
    unsigned short* __restrict__ wA,    // [B_][COUT][KDIM] bf16
    float* __restrict__ beff)           // [B_][COUT]
{
    __shared__ float w[NB];
    const int b = blockIdx.x, t = threadIdx.x;
    if (t < NB) w[t] = weights[b * NB + t];
    __syncthreads();

    const int E = COUT * KDIM;          // 36864, k = ci*9+kh*3+kw (natural order)
    for (int e = t; e < E; e += 256) {
        float acc = 0.f;
#pragma unroll
        for (int n = 0; n < NB; ++n) acc += w[n] * conv_w[n * E + e];
        wA[(size_t)b * E + e] = f2bf(acc);
    }
    if (t < COUT) {
        float acc = 0.f;
#pragma unroll
        for (int n = 0; n < NB; ++n) acc += w[n] * conv_b[n * COUT + t];
        beff[b * COUT + t] = acc;
    }
}

// ---- stage one K-slab (A: 64x32 bf16 weights; B: 64x32 bf16 im2col) -------
__device__ __forceinline__ void stage_slab(
    unsigned short* __restrict__ sA, unsigned short* __restrict__ sB,
    const unsigned short* __restrict__ Asrc_k,   // Abase + a_m*KDIM + a_c + k0
    const float* __restrict__ xb,
    int a_lds_off, int kkb, int ci, int r9, int lane, int xoff0, int xoff1)
{
    // A: one 16B vector per thread, global -> LDS
    *(uint4*)(sA + a_lds_off) = *(const uint4*)(Asrc_k);
    // B: 4 consecutive k per wave, 64 n across lanes (coalesced reads)
#pragma unroll
    for (int kk4 = 0; kk4 < 4; ++kk4) {
        const int kh = r9 / 3, kw = r9 - kh * 3;
        const float* xp = xb + ci * (H_ * W_) + kh * W_ + kw;
        sB[lane * LDSTR + kkb + kk4]        = f2bf(xp[xoff0]);
        sB[(lane + 32) * LDSTR + kkb + kk4] = f2bf(xp[xoff1]);
        if (++r9 == 9) { r9 = 0; ++ci; }
    }
}

// --------- kernel 3: per-batch implicit-GEMM 3x3 conv via WMMA bf16 --------
// M=COUT=64, N=3136 (tiles of 64), K=576 (slabs of 32), double-buffered LDS.
__global__ __launch_bounds__(256) void conv_wmma_kernel(
    const float* __restrict__ x,            // [B_][CIN][H_][W_]
    const unsigned short* __restrict__ wA,  // [B_][COUT][KDIM] bf16
    const float* __restrict__ beff,         // [B_][COUT]
    float* __restrict__ out)                // [B_][COUT][OH][OW]
{
    __shared__ alignas(16) unsigned short lA[2][64 * LDSTR];
    __shared__ alignas(16) unsigned short lB[2][64 * LDSTR];

    const int b  = blockIdx.x;              // batch
    const int N0 = blockIdx.y * 64;         // output-pixel tile base (49 tiles)
    const int t  = threadIdx.x;
    const int lane = t & 31, wave = t >> 5;

    const int mt  = wave & 3;               // M tile (16 rows)
    const int ntp = (wave >> 2) * 2;        // first of two N tiles

    v8f acc0 = {}; v8f acc1 = {};

    const unsigned short* Abase = wA + (size_t)b * COUT * KDIM;
    const float* xb = x + (size_t)b * CIN * H_ * W_;

    // A staging coords: 64 rows x 32 k, one 16B vector per thread
    const int a_m = t >> 2;
    const int a_c = (t & 3) * 8;
    const unsigned short* Asrc = Abase + a_m * KDIM + a_c;
    const int a_lds = a_m * LDSTR + a_c;

    // B staging coords (K-invariant): this thread's two output pixels
    const int kkb = wave * 4;               // this wave's 4 k's within slab
    const int n0g = N0 + lane;
    const int n1g = N0 + lane + 32;
    const int xoff0 = (n0g / OW) * W_ + (n0g % OW);
    const int xoff1 = (n1g / OW) * W_ + (n1g % OW);

    // fragment coords
    const int frl = lane & 15;              // row within 16x16 tile
    const int fh  = lane >> 4;              // lane half -> K half selector

    // incremental k -> (ci, r9) tracking for this wave's k base
    int ci_s = kkb / 9, r9_s = kkb % 9;

    // prologue: stage slab 0 into buffer 0
    stage_slab(lA[0], lB[0], Asrc, xb, a_lds, kkb, ci_s, r9_s, lane, xoff0, xoff1);
    r9_s += 5; ci_s += 3; if (r9_s >= 9) { r9_s -= 9; ++ci_s; }   // advance k by 32
    __syncthreads();

#pragma unroll 2
    for (int k0 = 0; k0 < KDIM; k0 += 32) {
        const int buf = (k0 >> 5) & 1;
        // stage next slab into the other buffer while computing this one
        if (k0 + 32 < KDIM) {
            stage_slab(lA[buf ^ 1], lB[buf ^ 1], Asrc + k0 + 32, xb, a_lds,
                       kkb, ci_s, r9_s, lane, xoff0, xoff1);
            r9_s += 5; ci_s += 3; if (r9_s >= 9) { r9_s -= 9; ++ci_s; }
            if (k0 + 64 < KDIM)
                __builtin_prefetch(Asrc + k0 + 64, 0, 0);   // global_prefetch_b8
        }

        // ---- fragments + WMMA on current buffer ----
        v16bf af = load_frag(&lA[buf][(mt * 16 + frl) * LDSTR + fh * 8]);
        v16bf bf0 = load_frag(&lB[buf][(ntp * 16 + frl) * LDSTR + fh * 8]);
        v16bf bf1 = load_frag(&lB[buf][((ntp + 1) * 16 + frl) * LDSTR + fh * 8]);

        acc0 = __builtin_amdgcn_wmma_f32_16x16x32_bf16(
            false, af, false, bf0, (short)0, acc0, false, false);
        acc1 = __builtin_amdgcn_wmma_f32_16x16x32_bf16(
            false, af, false, bf1, (short)0, acc1, false, false);
        __syncthreads();
    }

    // ---- epilogue: bias + store (softmax weights sum to 1 => /d is identity)
    const int nc0 = N0 + ntp * 16 + frl;
    const int nc1 = N0 + (ntp + 1) * 16 + frl;
#pragma unroll
    for (int v = 0; v < 8; ++v) {
        const int co = mt * 16 + fh * 8 + v;        // D layout: m = v + 8*(lane/16)
        const float bias = beff[b * COUT + co];
        float* orow = out + ((size_t)b * COUT + co) * NPIX;
        orow[nc0] = acc0[v] + bias;
        orow[nc1] = acc1[v] + bias;
    }
}

// ---------------------------------------------------------------------------
extern "C" void kernel_launch(void* const* d_in, const int* in_sizes, int n_in,
                              void* d_out, int out_size, void* d_ws, size_t ws_size,
                              hipStream_t stream) {
    const float* x   = (const float*)d_in[0];
    const float* rv  = (const float*)d_in[1];
    const float* W1  = (const float*)d_in[2];
    const float* b1  = (const float*)d_in[3];
    const float* W2  = (const float*)d_in[4];
    const float* b2  = (const float*)d_in[5];
    const float* emb = (const float*)d_in[6];
    const float* cw  = (const float*)d_in[7];
    const float* cb  = (const float*)d_in[8];
    float* out = (float*)d_out;

    char* ws = (char*)d_ws;
    float*          weights = (float*)(ws);                 // 32*10*4   = 1280 B
    float*          beff    = (float*)(ws + 2048);          // 32*64*4   = 8192 B
    unsigned short* wA      = (unsigned short*)(ws + 2048 + 8192); // 32*64*576*2 = 2.36 MB

    routing_kernel<<<B_, 128, 0, stream>>>(rv, W1, b1, W2, b2, emb, weights);
    mix_kernel<<<B_, 256, 0, stream>>>(cw, cb, weights, wA, beff);
    dim3 grid(B_, NPIX / 64);   // 32 x 49
    conv_wmma_kernel<<<grid, 256, 0, stream>>>(x, wA, beff, out);
}
